// HopeBlockWithAttention_89687507076235
// MI455X (gfx1250) — compile-verified, matrix operand order checked by reading, not run on hardware
//
#include <hip/hip_runtime.h>
#include <hip/hip_bf16.h>
#include <math.h>
#include <stdint.h>

// ---------------- problem constants ----------------
#define BSZ   2
#define SEQ   2048
#define DIMD  768
#define HEADS 12
#define HDH   64
#define WIN   512
#define LEV   4
#define EXPF  4
#define LRC   0.1f

#define BS    (BSZ * SEQ)          // 4096
#define BSD   ((size_t)BS * DIMD)  // 3,145,728 floats

typedef __attribute__((ext_vector_type(16))) __bf16 v16bf;
typedef __attribute__((ext_vector_type(8)))  float  v8f;

// optional gfx1250 async global->LDS path (probe via __has_builtin; fallback keeps compiling)
#if defined(__has_builtin)
#if __has_builtin(__builtin_amdgcn_global_load_async_to_lds_b32) && \
    __has_builtin(__builtin_amdgcn_s_wait_asynccnt)
#define HAVE_ASYNC_LDS 1
#endif
#endif

#if defined(HAVE_ASYNC_LDS)
typedef __attribute__((address_space(1))) int g1int_t;   // global
typedef __attribute__((address_space(3))) int l3int_t;   // LDS
#endif

// ---------------- helpers ----------------
// pack two fp32 -> bf16x2 dword: one integer round-add per value + one v_perm_b32
__device__ __forceinline__ unsigned pack_bf16x2(float f0, float f1) {
    union { float f; unsigned u; } a, b;
    a.f = f0; b.f = f1;
    unsigned ua = a.u + 0x8000u;
    unsigned ub = b.u + 0x8000u;
    // bytes: {ub[3],ub[2],ua[3],ua[2]} -> {bf16(f1), bf16(f0)}
    return __builtin_amdgcn_perm(ub, ua, 0x07060302u);
}

__device__ __forceinline__ __bf16 u16bf(unsigned short s) {
    union { unsigned short s; __bf16 b; } u; u.s = s; return u.b;
}

__device__ __forceinline__ float wave_sum(float v) {
    #pragma unroll
    for (int off = 16; off; off >>= 1) v += __shfl_xor(v, off, 32);
    return v;
}
__device__ __forceinline__ float wave_max(float v) {
    #pragma unroll
    for (int off = 16; off; off >>= 1) v = fmaxf(v, __shfl_xor(v, off, 32));
    return v;
}

// ---------------- generic bf16 WMMA GEMM ----------------
// C[M,N] = epilogue( sum_k A[m,k] * B[n,k] ), A:[M,K] lda, B:[N,K] ldb (row-major, K contiguous)
// flags: 1=add bias[n], 2=gelu(tanh), 4=accumulate into existing C;  val = alpha*(acc+bias[gelu]) (+Cold)
#define GBM 128
#define GBN 128
#define GBK 32
#define GLS 34   // padded LDS row stride (u16 elements) -> 4B-aligned dword pairs

__global__ __launch_bounds__(256) void gemm_bf16_wmma(
    const float* __restrict__ A, int lda,
    const float* __restrict__ Bw, int ldb,
    const float* __restrict__ bias,
    float* __restrict__ C, int ldc,
    int M, int N, int K, int flags, float alpha)
{
    __shared__ unsigned short As[GBM * GLS];
    __shared__ unsigned short Bs[GBN * GLS];

    const int tid  = threadIdx.x;
    const int wave = tid >> 5;
    const int lane = tid & 31;
    const int m0 = blockIdx.x * GBM;
    const int n0 = blockIdx.y * GBN;

    v8f acc[8];
    #pragma unroll
    for (int g = 0; g < 8; ++g) { v8f z = {}; acc[g] = z; }

    // staging map: 2 threads per row, 16 consecutive floats each (16B-aligned float4 loads)
    const int sr  = tid >> 1;
    const int scb = (tid & 1) * 16;

    for (int k0 = 0; k0 < K; k0 += GBK) {
        // stage A tile 128x32 -> bf16x2 packed dword stores
        {
            const float* src = A + (size_t)(m0 + sr) * lda + k0 + scb;
            if (k0 + GBK < K) __builtin_prefetch(src + GBK, 0, 3);
            const float4* s4 = (const float4*)src;
            unsigned* dst = (unsigned*)(As + sr * GLS + scb);
            #pragma unroll
            for (int i = 0; i < 4; ++i) {
                float4 v = s4[i];
                dst[2 * i]     = pack_bf16x2(v.x, v.y);
                dst[2 * i + 1] = pack_bf16x2(v.z, v.w);
            }
        }
        // stage B tile 128x32
        {
            const float* src = Bw + (size_t)(n0 + sr) * ldb + k0 + scb;
            if (k0 + GBK < K) __builtin_prefetch(src + GBK, 0, 3);
            const float4* s4 = (const float4*)src;
            unsigned* dst = (unsigned*)(Bs + sr * GLS + scb);
            #pragma unroll
            for (int i = 0; i < 4; ++i) {
                float4 v = s4[i];
                dst[2 * i]     = pack_bf16x2(v.x, v.y);
                dst[2 * i + 1] = pack_bf16x2(v.z, v.w);
            }
        }
        __syncthreads();

        // A fragment (16x32), ISA 16-bit A layout:
        // lane<16: row=lane,   K {0..7,16..23}; lane>=16: row=lane-16, K {8..15,24..31}
        const int half = lane >> 4;
        const int mr   = lane & 15;
        const unsigned short* arow = As + (wave * 16 + mr) * GLS;
        v16bf afrag;
        #pragma unroll
        for (int i = 0; i < 8; ++i) afrag[i]     = u16bf(arow[i + half * 8]);
        #pragma unroll
        for (int i = 0; i < 8; ++i) afrag[8 + i] = u16bf(arow[16 + i + half * 8]);

        #pragma unroll
        for (int g = 0; g < 8; ++g) {
            const unsigned short* brow = Bs + (g * 16 + mr) * GLS;
            v16bf bfrag;
            #pragma unroll
            for (int i = 0; i < 8; ++i) bfrag[i]     = u16bf(brow[i + half * 8]);
            #pragma unroll
            for (int i = 0; i < 8; ++i) bfrag[8 + i] = u16bf(brow[16 + i + half * 8]);
            acc[g] = __builtin_amdgcn_wmma_f32_16x16x32_bf16(
                false, afrag, false, bfrag, (short)0, acc[g], false, false);
        }
        __syncthreads();
    }

    // epilogue: C/D layout — VGPR r: lane<16 -> M=r, lane>=16 -> M=8+r; N = lane&15 (+16*g)
    const int half = lane >> 4;
    const int nn   = lane & 15;
    #pragma unroll
    for (int g = 0; g < 8; ++g) {
        const int col = n0 + g * 16 + nn;
        const float bv = (flags & 1) ? bias[col] : 0.f;
        #pragma unroll
        for (int r = 0; r < 8; ++r) {
            const int row = m0 + wave * 16 + half * 8 + r;
            float v = acc[g][r] + bv;
            if (flags & 2) {                               // tanh-approx GELU
                float x3 = v * v * v;
                v = 0.5f * v * (1.f + tanhf(0.7978845608f * (v + 0.044715f * x3)));
            }
            v *= alpha;
            const size_t ci = (size_t)row * ldc + col;
            if (flags & 4) v += C[ci];
            C[ci] = v;
        }
    }
}

// ---------------- sliding-window causal attention ----------------
// qkv: [BS, 3*D] rows; one wave per (b,h,i) query row; 4 waves / block.
__global__ __launch_bounds__(128) void attn_window_kernel(
    const float* __restrict__ qkv, float* __restrict__ o)
{
    __shared__ float sq[4][HDH];
    __shared__ float sc[4][WIN];

    const int wslot = threadIdx.x >> 5;
    const int lane  = threadIdx.x & 31;
    const int wid   = blockIdx.x * 4 + wslot;     // grid sized exactly
    const int i  = wid % SEQ;
    const int bh = wid / SEQ;
    const int h  = bh % HEADS;
    const int b  = bh / HEADS;

    const int RS = 3 * DIMD;
    const float scale = 0.125f;                   // 1/sqrt(64)
    const float* qrow = qkv + (size_t)(b * SEQ + i) * RS + h * HDH;
    sq[wslot][2 * lane]     = qrow[2 * lane] * scale;
    sq[wslot][2 * lane + 1] = qrow[2 * lane + 1] * scale;
    __syncthreads();

    int j0  = i - (WIN - 1); if (j0 < 0) j0 = 0;
    const int cnt = i - j0 + 1;

    // pass 1: scores (lane-per-key)
    float lmax = -1e30f;
    for (int jj = lane; jj < cnt; jj += 32) {
        const float* krow = qkv + (size_t)(b * SEQ + j0 + jj) * RS + DIMD + h * HDH;
        float s = 0.f;
        #pragma unroll
        for (int d = 0; d < HDH; d += 4) {
            s += sq[wslot][d]     * krow[d];
            s += sq[wslot][d + 1] * krow[d + 1];
            s += sq[wslot][d + 2] * krow[d + 2];
            s += sq[wslot][d + 3] * krow[d + 3];
        }
        sc[wslot][jj] = s;
        lmax = fmaxf(lmax, s);
    }
    const float m = wave_max(lmax);

    float lsum = 0.f;
    for (int jj = lane; jj < cnt; jj += 32) {
        float e = __expf(sc[wslot][jj] - m);
        sc[wslot][jj] = e;
        lsum += e;
    }
    const float inv = 1.f / wave_sum(lsum);

    // pass 2: o = sum_j p_j * v_j  (2 dims per lane, coalesced v loads)
    float o0 = 0.f, o1 = 0.f;
    for (int j = 0; j < cnt; ++j) {
        const float p = sc[wslot][j] * inv;
        const float* vrow = qkv + (size_t)(b * SEQ + j0 + j) * RS + 2 * DIMD + h * HDH;
        o0 += p * vrow[2 * lane];
        o1 += p * vrow[2 * lane + 1];
    }
    float* orow = o + (size_t)(b * SEQ + i) * DIMD + h * HDH;
    orow[2 * lane]     = o0;
    orow[2 * lane + 1] = o1;
}

// ---------------- delta-rule recurrent memory (sequential scan) ----------------
// one block per (b,h), 64 threads; thread e owns column M[:,e] in registers.
__global__ __launch_bounds__(64) void delta_scan_kernel(
    const float* __restrict__ tq, const float* __restrict__ tk,
    const float* __restrict__ tv, float* __restrict__ oseq,
    float* __restrict__ mstate)
{
    const int bh = blockIdx.x;
    const int h  = bh % HEADS;
    const int b  = bh / HEADS;
    const int e  = threadIdx.x;

    float Mcol[HDH];
    #pragma unroll
    for (int d = 0; d < HDH; ++d) Mcol[d] = 0.f;

    __shared__ float sk[HDH], sqv[HDH];

    for (int t = 0; t < SEQ; ++t) {
        const size_t base = (size_t)(b * SEQ + t) * DIMD + h * HDH;
#if defined(HAVE_ASYNC_LDS)
        // gfx1250 async global->LDS broadcast of k_t and q_t rows
        __builtin_amdgcn_global_load_async_to_lds_b32(
            (g1int_t*)(uintptr_t)(tk + base + e),
            (l3int_t*)(unsigned)(uintptr_t)(&sk[e]),
            0, 0);
        __builtin_amdgcn_global_load_async_to_lds_b32(
            (g1int_t*)(uintptr_t)(tq + base + e),
            (l3int_t*)(unsigned)(uintptr_t)(&sqv[e]),
            0, 0);
        const float ve = tv[base + e];
        __builtin_amdgcn_s_wait_asynccnt(0);
#else
        sk[e]  = tk[base + e];
        sqv[e] = tq[base + e];
        const float ve = tv[base + e];
#endif
        __syncthreads();

        float pred = 0.f;
        #pragma unroll
        for (int d = 0; d < HDH; ++d) pred += Mcol[d] * sk[d];
        const float dl = LRC * (ve - pred);

        float ot = 0.f;
        #pragma unroll
        for (int d = 0; d < HDH; ++d) {
            Mcol[d] += sk[d] * dl;
            ot += Mcol[d] * sqv[d];
        }
        oseq[base + e] = ot;
        __syncthreads();
    }
    #pragma unroll
    for (int d = 0; d < HDH; ++d)
        mstate[((size_t)bh * HDH + d) * HDH + e] = Mcol[d];   // [B,H,dh,dh]
}

// ---------------- layernorm with residual: out = LN(x + res) * g + b ----------------
__global__ __launch_bounds__(256) void ln_res_kernel(
    const float* __restrict__ xb, const float* __restrict__ res,
    const float* __restrict__ g, const float* __restrict__ bta,
    float* __restrict__ out)
{
    const int row = blockIdx.x;
    const int tid = threadIdx.x;
    const size_t ro = (size_t)row * DIMD;
    const float v0 = xb[ro + tid]        + res[ro + tid];
    const float v1 = xb[ro + tid + 256]  + res[ro + tid + 256];
    const float v2 = xb[ro + tid + 512]  + res[ro + tid + 512];

    float s  = v0 + v1 + v2;
    float sq = v0 * v0 + v1 * v1 + v2 * v2;
    __shared__ float rs[8], rq[8];
    const float ws = wave_sum(s), wq = wave_sum(sq);
    const int wave = tid >> 5, lane = tid & 31;
    if (!lane) { rs[wave] = ws; rq[wave] = wq; }
    __syncthreads();
    float tS = 0.f, tQ = 0.f;
    #pragma unroll
    for (int i = 0; i < 8; ++i) { tS += rs[i]; tQ += rq[i]; }
    const float mean = tS * (1.f / DIMD);
    const float var  = tQ * (1.f / DIMD) - mean * mean;
    const float rinv = rsqrtf(var + 1e-5f);

    out[ro + tid]       = (v0 - mean) * rinv * g[tid]       + bta[tid];
    out[ro + tid + 256] = (v1 - mean) * rinv * g[tid + 256] + bta[tid + 256];
    out[ro + tid + 512] = (v2 - mean) * rinv * g[tid + 512] + bta[tid + 512];
}

// ---------------- gated combine: xc = sig(gpre)*xl + (1-sig)*xg (in place) ----------------
__global__ __launch_bounds__(256) void gate_combine_kernel(
    float* __restrict__ gbuf, const float* __restrict__ xl,
    const float* __restrict__ xg)
{
    const size_t i = (size_t)blockIdx.x * 256 + threadIdx.x;
    const float sgm = 1.f / (1.f + __expf(-gbuf[i]));
    gbuf[i] = sgm * xl[i] + (1.f - sgm) * xg[i];
}

// ---------------- launcher ----------------
extern "C" void kernel_launch(void* const* d_in, const int* in_sizes, int n_in,
                              void* d_out, int out_size, void* d_ws, size_t ws_size,
                              hipStream_t stream) {
    (void)in_sizes; (void)n_in; (void)out_size; (void)ws_size;

    const float* x          = (const float*)d_in[0];
    const float* in_proj_w  = (const float*)d_in[1];
    const float* in_proj_b  = (const float*)d_in[2];
    const float* out_proj_w = (const float*)d_in[3];
    const float* out_proj_b = (const float*)d_in[4];
    const float* ln_local_g = (const float*)d_in[5];
    const float* ln_local_b = (const float*)d_in[6];
    const float* t_wq       = (const float*)d_in[7];
    const float* t_wk       = (const float*)d_in[8];
    const float* t_wv       = (const float*)d_in[9];
    const float* t_wo       = (const float*)d_in[10];
    const float* ln_tit_g   = (const float*)d_in[11];
    const float* ln_tit_b   = (const float*)d_in[12];
    const float* gate_w     = (const float*)d_in[13];
    const float* gate_b     = (const float*)d_in[14];
    const float* cms_w1     = (const float*)d_in[15];
    const float* cms_b1     = (const float*)d_in[16];
    const float* cms_w2     = (const float*)d_in[17];
    const float* cms_b2     = (const float*)d_in[18];
    const float* ln_cms_g   = (const float*)d_in[19];
    const float* ln_cms_b   = (const float*)d_in[20];

    float* outp   = (float*)d_out;            // [B,S,D]
    float* mstate = outp + BSD;               // [B,H,dh,dh]

    // workspace layout (floats): 12*BSD total (~151 MB)
    float* ws      = (float*)d_ws;
    float* qkvbuf  = ws;                      // 3*BSD : qkv, later tq/tk/tv
    float* tmp1    = ws + 3 * BSD;            // BSD   : local_out / titans_out / cms accum
    float* x_local = ws + 4 * BSD;            // BSD
    float* x_glob  = ws + 5 * BSD;            // BSD
    float* gbuf    = ws + 6 * BSD;            // BSD   : gate pre-act -> xc
    float* oseq    = ws + 7 * BSD;            // BSD   : attn o, then titans o_seq
    float* hbuf    = ws + 8 * BSD;            // 4*BSD : CMS hidden [BS, 3072]

    const dim3 blk(256);
    const int N3D = 3 * DIMD, NE = EXPF * DIMD;

    // 1) qkv = x @ in_proj_w.T + b
    gemm_bf16_wmma<<<dim3(BS / GBM, N3D / GBN), blk, 0, stream>>>(
        x, DIMD, in_proj_w, DIMD, in_proj_b, qkvbuf, N3D, BS, N3D, DIMD, 1, 1.f);
    // 2) sliding-window attention -> oseq
    attn_window_kernel<<<dim3(BSZ * HEADS * SEQ / 4), dim3(128), 0, stream>>>(qkvbuf, oseq);
    // 3) local_out = o @ out_proj_w.T + b
    gemm_bf16_wmma<<<dim3(BS / GBM, DIMD / GBN), blk, 0, stream>>>(
        oseq, DIMD, out_proj_w, DIMD, out_proj_b, tmp1, DIMD, BS, DIMD, DIMD, 1, 1.f);
    // 4) x_local = LN(x + local_out)
    ln_res_kernel<<<dim3(BS), blk, 0, stream>>>(x, tmp1, ln_local_g, ln_local_b, x_local);

    // 5) titans projections (reuse qkv buffer)
    float* tq = qkvbuf; float* tk = qkvbuf + BSD; float* tv = qkvbuf + 2 * BSD;
    gemm_bf16_wmma<<<dim3(BS / GBM, DIMD / GBN), blk, 0, stream>>>(
        x, DIMD, t_wq, DIMD, nullptr, tq, DIMD, BS, DIMD, DIMD, 0, 1.f);
    gemm_bf16_wmma<<<dim3(BS / GBM, DIMD / GBN), blk, 0, stream>>>(
        x, DIMD, t_wk, DIMD, nullptr, tk, DIMD, BS, DIMD, DIMD, 0, 1.f);
    gemm_bf16_wmma<<<dim3(BS / GBM, DIMD / GBN), blk, 0, stream>>>(
        x, DIMD, t_wv, DIMD, nullptr, tv, DIMD, BS, DIMD, DIMD, 0, 1.f);
    // 6) delta-rule scan -> oseq, final M -> d_out tail
    delta_scan_kernel<<<dim3(BSZ * HEADS), dim3(64), 0, stream>>>(tq, tk, tv, oseq, mstate);
    // 7) titans_out = o_seq @ t_wo.T
    gemm_bf16_wmma<<<dim3(BS / GBM, DIMD / GBN), blk, 0, stream>>>(
        oseq, DIMD, t_wo, DIMD, nullptr, tmp1, DIMD, BS, DIMD, DIMD, 0, 1.f);
    // 8) x_global = LN(x + titans_out)
    ln_res_kernel<<<dim3(BS), blk, 0, stream>>>(x, tmp1, ln_tit_g, ln_tit_b, x_glob);

    // 9) gate pre-act: gbuf = x_local @ gate_w[:, :D].T ; += x_global @ gate_w[:, D:].T + b
    gemm_bf16_wmma<<<dim3(BS / GBM, DIMD / GBN), blk, 0, stream>>>(
        x_local, DIMD, gate_w, 2 * DIMD, nullptr, gbuf, DIMD, BS, DIMD, DIMD, 0, 1.f);
    gemm_bf16_wmma<<<dim3(BS / GBM, DIMD / GBN), blk, 0, stream>>>(
        x_glob, DIMD, gate_w + DIMD, 2 * DIMD, gate_b, gbuf, DIMD, BS, DIMD, DIMD, 1 | 4, 1.f);
    // 10) xc = sig(gbuf)*x_local + (1-sig)*x_global  (in place)
    gate_combine_kernel<<<dim3((unsigned)(BSD / 256)), blk, 0, stream>>>(gbuf, x_local, x_glob);

    // 11) CMS: mean over 4 FFN levels, accumulated into tmp1 with alpha = 1/4
    for (int l = 0; l < LEV; ++l) {
        gemm_bf16_wmma<<<dim3(BS / GBM, NE / GBN), blk, 0, stream>>>(
            gbuf, DIMD, cms_w1 + (size_t)l * NE * DIMD, DIMD,
            cms_b1 + (size_t)l * NE, hbuf, NE, BS, NE, DIMD, 1 | 2, 1.f);
        gemm_bf16_wmma<<<dim3(BS / GBM, DIMD / GBN), blk, 0, stream>>>(
            hbuf, NE, cms_w2 + (size_t)l * DIMD * NE, NE,
            cms_b2 + (size_t)l * DIMD, tmp1, DIMD, BS, DIMD, NE,
            1 | (l ? 4 : 0), 0.25f);
    }
    // 12) out = LN(xc + cms_mean)
    ln_res_kernel<<<dim3(BS), blk, 0, stream>>>(gbuf, tmp1, ln_cms_g, ln_cms_b, outp);
}
